// Gemma2Attention_10823317586073
// MI455X (gfx1250) — matrix operand order repaired and checked by reading, not compiled, not run on hardware
//
#include <hip/hip_runtime.h>
#include <hip/hip_bf16.h>

typedef __attribute__((ext_vector_type(16))) __bf16 v16bf;
typedef __attribute__((ext_vector_type(8)))  float  v8f;

#define S_LEN   4096
#define HID_DIM 2304
#define N_H     8
#define N_KV    4
#define H_D     256
#define SWIN    2048
#define QDIM    (N_H * H_D)    // 2048
#define KVDIM   (N_KV * H_D)   // 1024

// ---------------------------------------------------------------- CDNA5 helpers

__device__ __forceinline__ v8f wmma_bf16(v16bf a, v16bf b, v8f c) {
  // 8 args: (neg_a, A, neg_b, B, c_mod, C, reuse_a, reuse_b)
  return __builtin_amdgcn_wmma_f32_16x16x32_bf16(false, a, false, b, (short)0, c,
                                                 false, false);
}

// Async DMA: global -> LDS, 16 bytes per lane, tracked by ASYNCcnt.
__device__ __forceinline__ void async_to_lds_b128(unsigned lds_addr, const void* gaddr) {
  asm volatile("global_load_async_to_lds_b128 %0, %1, off"
               :: "v"(lds_addr), "v"(gaddr) : "memory");
}
__device__ __forceinline__ void wait_async0() {
  asm volatile("s_wait_asynccnt 0" ::: "memory");
}
// Low 32 bits of a generic LDS pointer = LDS byte address.
__device__ __forceinline__ unsigned lds_addr_of(const void* p) {
  return (unsigned)(size_t)p;
}

// A-matrix 16x32 bf16 frag: lanes 0-15 hold row=lane, K {0..7,16..23};
// lanes 16-31 hold row=lane-16, K {8..15,24..31}.
__device__ __forceinline__ v16bf load_frag_a(const __bf16* base, int stride, int lane) {
  int row = lane & 15;
  int kh  = (lane >> 4) << 3;  // 0 or 8
  const __bf16* p = base + row * stride;
  v16bf f;
  uint4* fp = reinterpret_cast<uint4*>(&f);
  fp[0] = *reinterpret_cast<const uint4*>(p + kh);
  fp[1] = *reinterpret_cast<const uint4*>(p + 16 + kh);
  return f;
}

// B-matrix 32x16 bf16 frag from an [N][K] tile: lane holds col N=lane%16,
// K {0..7,16..23} (lanes 0-15) or {8..15,24..31} (lanes 16-31).
__device__ __forceinline__ v16bf load_frag_b(const __bf16* base, int stride, int lane) {
  int n  = lane & 15;
  int kh = (lane >> 4) << 3;   // 0 or 8
  const __bf16* p = base + n * stride + kh;
  v16bf f;
  uint4* fp = reinterpret_cast<uint4*>(&f);
  fp[0] = *reinterpret_cast<const uint4*>(p);
  fp[1] = *reinterpret_cast<const uint4*>(p + 16);
  return f;
}

__device__ __forceinline__ float softcap50(float x) {
  float t = x * 0.02f;                         // /50
  t = fminf(fmaxf(t, -15.f), 15.f);
  float e = __expf(2.f * t);
  return 50.f * ((e - 1.f) / (e + 1.f));       // 50*tanh(x/50)
}

// ---------------------------------------------------------------- elementwise

__global__ void cvt_kernel(const float* __restrict__ in, __bf16* __restrict__ out, int n) {
  int i = blockIdx.x * blockDim.x + threadIdx.x;
  if (i < n) out[i] = (__bf16)in[i];
}

// In-place RoPE on bf16 activations laid out [s][nheads*256].
__global__ void rope_kernel(__bf16* __restrict__ X, int nheads) {
  int idx = blockIdx.x * blockDim.x + threadIdx.x;
  int total = S_LEN * nheads * 128;
  if (idx >= total) return;
  int d  = idx & 127;
  int t  = idx >> 7;
  int hh = t % nheads;
  int s  = t / nheads;
  float inv = __powf(10000.0f, -(float)d * (1.0f / 128.0f));
  float fr  = (float)s * inv;
  float c = __cosf(fr), sn = __sinf(fr);
  size_t base = (size_t)s * nheads * H_D + (size_t)hh * H_D;
  float x1 = (float)X[base + d];
  float x2 = (float)X[base + d + 128];
  X[base + d]       = (__bf16)(x1 * c - x2 * sn);
  X[base + d + 128] = (__bf16)(x2 * c + x1 * sn);
}

// ---------------------------------------------------------------- GEMM Y = X * W^T
// A: [M][K] bf16 row-major, B(=W): [N][K] bf16 row-major, C: [M][N].
// 128x128 block tile, BK=32, 8 waves; double-buffered LDS filled by async DMA.
template <bool OUT_BF16>
__global__ __launch_bounds__(256) void gemm_xwt(
    const __bf16* __restrict__ A, const __bf16* __restrict__ B,
    void* __restrict__ Cout, int M, int N, int Kd)
{
  __shared__ __bf16 sA[2][128 * 40];   // 32 K + 8 pad halves (80B row, 16B aligned)
  __shared__ __bf16 sB[2][128 * 40];
  const int tid  = threadIdx.x;
  const int wave = tid >> 5, lane = tid & 31;
  const int ln = lane & 15, lh = lane >> 4;
  const int mBase = blockIdx.y * 128;
  const int nBase = blockIdx.x * 128;
  const int wm = (wave >> 2) * 64;  // 0 or 64
  const int wn = (wave & 3) * 32;   // 0,32,64,96

  v8f zero = {0, 0, 0, 0, 0, 0, 0, 0};
  v8f acc[4][2];
  for (int mi = 0; mi < 4; ++mi)
    for (int ni = 0; ni < 2; ++ni) acc[mi][ni] = zero;

  auto fill = [&](int buf, int k0) {
    #pragma unroll
    for (int i = 0; i < 2; ++i) {
      int slot = tid + (i << 8);       // 512 x 16B chunks per matrix tile
      int r = slot >> 2, ch = slot & 3;
      async_to_lds_b128(lds_addr_of(&sA[buf][r * 40 + ch * 8]),
                        A + (size_t)(mBase + r) * Kd + k0 + ch * 8);
      async_to_lds_b128(lds_addr_of(&sB[buf][r * 40 + ch * 8]),
                        B + (size_t)(nBase + r) * Kd + k0 + ch * 8);
    }
  };

  const int nk = Kd >> 5;
  fill(0, 0);
  wait_async0();
  __syncthreads();

  int buf = 0;
  for (int kt = 0; kt < nk; ++kt) {
    if (kt + 1 < nk) fill(buf ^ 1, (kt + 1) << 5);   // overlap DMA with WMMA

    v16bf b0 = load_frag_b(&sB[buf][(wn + 0)  * 40], 40, lane);
    v16bf b1 = load_frag_b(&sB[buf][(wn + 16) * 40], 40, lane);
    #pragma unroll
    for (int mi = 0; mi < 4; ++mi) {
      v16bf af = load_frag_a(&sA[buf][(wm + mi * 16) * 40], 40, lane);
      acc[mi][0] = wmma_bf16(af, b0, acc[mi][0]);
      acc[mi][1] = wmma_bf16(af, b1, acc[mi][1]);
    }
    wait_async0();
    __syncthreads();
    buf ^= 1;
  }

  // C layout: VGPR v -> row v (lanes 0-15) or v+8 (lanes 16-31), col = lane%16.
  #pragma unroll
  for (int mi = 0; mi < 4; ++mi)
    #pragma unroll
    for (int ni = 0; ni < 2; ++ni)
      #pragma unroll
      for (int v = 0; v < 8; ++v) {
        int r = mBase + wm + mi * 16 + v + (lh << 3);
        int c = nBase + wn + ni * 16 + ln;
        float val = acc[mi][ni][v];
        if (OUT_BF16)
          reinterpret_cast<__bf16*>(Cout)[(size_t)r * N + c] = (__bf16)val;
        else
          reinterpret_cast<float*>(Cout)[(size_t)r * N + c] = val;
      }
}

// ---------------------------------------------------------------- flash attention
// Grid: (S/64, NH).  128 threads = 4 waves; wave owns a 16-row q tile.
// K processed in 32-row blocks; double-buffered LDS, K filled by async DMA,
// V transposed through VGPRs, next-next block prefetched into cache.
__global__ __launch_bounds__(128) void flash_attn(
    const __bf16* __restrict__ Q, const __bf16* __restrict__ K,
    const __bf16* __restrict__ V, __bf16* __restrict__ O)
{
  __shared__ __bf16 sK[2][32 * 264];    // [k-row][dim], +8 pad halves
  __shared__ __bf16 sVt[2][256 * 32];   // [dim][k-row]
  __shared__ __bf16 sP[4][16 * 32];     // per-wave P staging [row][k-col]

  const int h    = blockIdx.y;
  const int q0   = blockIdx.x * 64;
  const int kvh  = h >> 1;              // GQA groups = 2
  const int tid  = threadIdx.x;
  const int wave = tid >> 5, lane = tid & 31;
  const int ln = lane & 15, lh = lane >> 4;
  const int m0 = q0 + wave * 16;

  // Q fragments (8 frags cover HD=256), kept in registers for the whole pass.
  v16bf qf[8];
  {
    const __bf16* qp = Q + (size_t)(m0 + ln) * QDIM + (size_t)h * H_D;
    #pragma unroll
    for (int kk = 0; kk < 8; ++kk) {
      const __bf16* p = qp + kk * 32;
      uint4* fp = reinterpret_cast<uint4*>(&qf[kk]);
      fp[0] = *reinterpret_cast<const uint4*>(p + (lh << 3));
      fp[1] = *reinterpret_cast<const uint4*>(p + 16 + (lh << 3));
    }
  }

  v8f zero = {0, 0, 0, 0, 0, 0, 0, 0};
  float m8[8], l8[8];
  v8f o[16];
  #pragma unroll
  for (int v = 0; v < 8; ++v) { m8[v] = -1e30f; l8[v] = 0.f; }
  #pragma unroll
  for (int nd = 0; nd < 16; ++nd) o[nd] = zero;

  auto fill_k = [&](int j0, int buf) {
    #pragma unroll
    for (int i = 0; i < 8; ++i) {
      int c = tid + (i << 7);           // 1024 x 16B chunks
      int r = c >> 5, ch = c & 31;
      async_to_lds_b128(lds_addr_of(&sK[buf][r * 264 + ch * 8]),
                        K + (size_t)(j0 + r) * KVDIM + (size_t)kvh * H_D + ch * 8);
    }
  };
  auto fill_v = [&](int j0, int buf) {
    #pragma unroll
    for (int i = 0; i < 8; ++i) {
      int c = tid + (i << 7);
      int r = c >> 5, ch = c & 31;
      uint4 d = *reinterpret_cast<const uint4*>(V + (size_t)(j0 + r) * KVDIM +
                                                (size_t)kvh * H_D + ch * 8);
      const __bf16* e = reinterpret_cast<const __bf16*>(&d);
      #pragma unroll
      for (int jj = 0; jj < 8; ++jj) sVt[buf][(ch * 8 + jj) * 32 + r] = e[jj];
    }
  };
  auto prefetch_kv = [&](int j0) {    // warm L2/WGP$ for block-after-next
    int r = tid >> 2, ch = tid & 3;   // 32 rows x 4 x 128B
    const __bf16* pv = V + (size_t)(j0 + r) * KVDIM + (size_t)kvh * H_D + ch * 64;
    const __bf16* pk = K + (size_t)(j0 + r) * KVDIM + (size_t)kvh * H_D + ch * 64;
    __builtin_prefetch(pv, 0, 1);
    __builtin_prefetch(pk, 0, 1);
  };

  int jmin = q0 - (SWIN - 1); if (jmin < 0) jmin = 0;
  const int kb_lo = jmin >> 5;
  const int kb_hi = (q0 + 63) >> 5;

  fill_k(kb_lo << 5, 0);
  fill_v(kb_lo << 5, 0);
  if (kb_lo + 1 <= kb_hi) prefetch_kv((kb_lo + 1) << 5);
  wait_async0();
  __syncthreads();

  int buf = 0;
  for (int kb = kb_lo; kb <= kb_hi; ++kb) {
    const int j0 = kb << 5;
    if (kb + 1 <= kb_hi) {
      fill_k((kb + 1) << 5, buf ^ 1);         // async DMA overlaps compute
      fill_v((kb + 1) << 5, buf ^ 1);
      if (kb + 2 <= kb_hi) prefetch_kv((kb + 2) << 5);
    }

    // S = Q * K^T   (16 x 32 per wave)
    v8f s0 = zero, s1 = zero;
    #pragma unroll
    for (int kk = 0; kk < 8; ++kk) {
      v16bf b0 = load_frag_b(&sK[buf][0  * 264 + kk * 32], 264, lane);
      v16bf b1 = load_frag_b(&sK[buf][16 * 264 + kk * 32], 264, lane);
      s0 = wmma_bf16(qf[kk], b0, s0);
      s1 = wmma_bf16(qf[kk], b1, s1);
    }

    // scale -> softcap -> causal+window mask -> online softmax update
    float mnew[8], corr[8], lp[8];
    #pragma unroll
    for (int v = 0; v < 8; ++v) {
      int i = m0 + v + (lh << 3);
      float x0 = softcap50(s0[v] * 0.0625f);
      float x1 = softcap50(s1[v] * 0.0625f);
      int ja = j0 + ln, jb = j0 + 16 + ln;
      if (!(ja <= i && (i - ja) < SWIN)) x0 = -1e30f;
      if (!(jb <= i && (i - jb) < SWIN)) x1 = -1e30f;
      s0[v] = x0; s1[v] = x1;
      mnew[v] = fmaxf(m8[v], fmaxf(x0, x1));
    }
    #pragma unroll
    for (int off = 1; off < 16; off <<= 1)
      #pragma unroll
      for (int v = 0; v < 8; ++v)
        mnew[v] = fmaxf(mnew[v], __shfl_xor(mnew[v], off, 32));
    #pragma unroll
    for (int v = 0; v < 8; ++v) {
      float mn = mnew[v];
      corr[v]  = __expf(m8[v] - mn);
      float p0 = __expf(s0[v] - mn);
      float p1 = __expf(s1[v] - mn);
      s0[v] = p0; s1[v] = p1;
      lp[v] = p0 + p1;
      m8[v] = mn;
    }
    #pragma unroll
    for (int off = 1; off < 16; off <<= 1)
      #pragma unroll
      for (int v = 0; v < 8; ++v)
        lp[v] += __shfl_xor(lp[v], off, 32);
    #pragma unroll
    for (int v = 0; v < 8; ++v) l8[v] = l8[v] * corr[v] + lp[v];

    #pragma unroll
    for (int nd = 0; nd < 16; ++nd)
      #pragma unroll
      for (int v = 0; v < 8; ++v) o[nd][v] *= corr[v];

    // stage P (16x32) to wave-private LDS; wave-local sync via dscnt only
    __bf16* pw = &sP[wave][0];
    #pragma unroll
    for (int v = 0; v < 8; ++v) {
      pw[(v + (lh << 3)) * 32 + ln]      = (__bf16)s0[v];
      pw[(v + (lh << 3)) * 32 + 16 + ln] = (__bf16)s1[v];
    }
    asm volatile("s_wait_dscnt 0" ::: "memory");
    v16bf pf = load_frag_a(pw, 32, lane);

    // O += P * V
    #pragma unroll
    for (int nd = 0; nd < 16; ++nd) {
      v16bf vf = load_frag_b(&sVt[buf][(nd << 4) * 32], 32, lane);
      o[nd] = wmma_bf16(pf, vf, o[nd]);
    }

    wait_async0();
    __syncthreads();
    buf ^= 1;
  }

  float linv[8];
  #pragma unroll
  for (int v = 0; v < 8; ++v) linv[v] = 1.0f / l8[v];
  #pragma unroll
  for (int nd = 0; nd < 16; ++nd)
    #pragma unroll
    for (int v = 0; v < 8; ++v) {
      int r = m0 + v + (lh << 3);
      int c = (nd << 4) + ln;
      O[(size_t)r * QDIM + (size_t)h * H_D + c] = (__bf16)(o[nd][v] * linv[v]);
    }
}

// ---------------------------------------------------------------- launch

extern "C" void kernel_launch(void* const* d_in, const int* in_sizes, int n_in,
                              void* d_out, int out_size, void* d_ws, size_t ws_size,
                              hipStream_t stream)
{
  const float* hs = (const float*)d_in[0];
  // d_in[1] = position_ids (arange; RoPE uses s directly)
  const float* Wq = (const float*)d_in[2];
  const float* Wk = (const float*)d_in[3];
  const float* Wv = (const float*)d_in[4];
  const float* Wo = (const float*)d_in[5];

  char* ws = (char*)d_ws;
  size_t off = 0;
  auto alloc = [&](size_t bytes) -> void* {
    void* p = ws + off;
    off += (bytes + 255) & ~(size_t)255;
    return p;
  };
  __bf16* Xb  = (__bf16*)alloc((size_t)S_LEN * HID_DIM * 2);
  __bf16* Wqb = (__bf16*)alloc((size_t)QDIM  * HID_DIM * 2);
  __bf16* Wkb = (__bf16*)alloc((size_t)KVDIM * HID_DIM * 2);
  __bf16* Wvb = (__bf16*)alloc((size_t)KVDIM * HID_DIM * 2);
  __bf16* Wob = (__bf16*)alloc((size_t)HID_DIM * QDIM * 2);
  __bf16* Qb  = (__bf16*)alloc((size_t)S_LEN * QDIM * 2);
  __bf16* Kb  = (__bf16*)alloc((size_t)S_LEN * KVDIM * 2);
  __bf16* Vb  = (__bf16*)alloc((size_t)S_LEN * KVDIM * 2);
  __bf16* Ob  = (__bf16*)alloc((size_t)S_LEN * QDIM * 2);

  auto cvt = [&](const float* src, __bf16* dst, int n) {
    cvt_kernel<<<(n + 255) / 256, 256, 0, stream>>>(src, dst, n);
  };
  cvt(hs, Xb, S_LEN * HID_DIM);
  cvt(Wq, Wqb, QDIM * HID_DIM);
  cvt(Wk, Wkb, KVDIM * HID_DIM);
  cvt(Wv, Wvb, KVDIM * HID_DIM);
  cvt(Wo, Wob, HID_DIM * QDIM);

  gemm_xwt<true><<<dim3(QDIM / 128,  S_LEN / 128), 256, 0, stream>>>(Xb, Wqb, Qb, S_LEN, QDIM,  HID_DIM);
  gemm_xwt<true><<<dim3(KVDIM / 128, S_LEN / 128), 256, 0, stream>>>(Xb, Wkb, Kb, S_LEN, KVDIM, HID_DIM);
  gemm_xwt<true><<<dim3(KVDIM / 128, S_LEN / 128), 256, 0, stream>>>(Xb, Wvb, Vb, S_LEN, KVDIM, HID_DIM);

  rope_kernel<<<(S_LEN * N_H  * 128 + 255) / 256, 256, 0, stream>>>(Qb, N_H);
  rope_kernel<<<(S_LEN * N_KV * 128 + 255) / 256, 256, 0, stream>>>(Kb, N_KV);

  flash_attn<<<dim3(S_LEN / 64, N_H), 128, 0, stream>>>(Qb, Kb, Vb, Ob);

  gemm_xwt<false><<<dim3(HID_DIM / 128, S_LEN / 128), 256, 0, stream>>>(Ob, Wob, d_out, S_LEN, HID_DIM, QDIM);
}
